// DynamicGRU_867583394173
// MI455X (gfx1250) — compile-verified, matrix operand order checked by reading, not run on hardware
//
#include <hip/hip_runtime.h>
#include <hip/hip_bf16.h>
#include <math.h>

// ---------------------------------------------------------------------------
// DynamicGRU on MI455X (gfx1250).  B=16, T=4096, D=256, N=256.
//   K0: pack weights to bf16 WMMA operand tiles.  Two fused regions:
//       WX = [Wx_gates(32ct) | Wx_cand(16ct)]  (48 col-tiles x 8 k-chunks)
//       WH = [Wh_gates(32ct) | Wh_cand(16ct)]  -> branchless tile addressing.
//   K1: X[65536,256] @ WX[256,768] + bias via v_wmma_f32_16x16x32_bf16;
//       gate half -> XG (fp32 scratch), cand half stashed into d_out rows.
//   K2: one 16-wave workgroup, 4096 sequential steps; h kept in LDS in both
//       fp32 (update math) and bf16 (WMMA A operand, converted once per
//       element per step); XG/Xc slabs staged coalesced into LDS each step;
//       2 barriers/step.
//   d_ws: [bf16 packed weights 768 KiB][XG fp32 134 MiB]
// ---------------------------------------------------------------------------

typedef __bf16 bf16_t;
typedef bf16_t v16bf __attribute__((ext_vector_type(16)));
typedef bf16_t v8bf  __attribute__((ext_vector_type(8)));
typedef float  v8f   __attribute__((ext_vector_type(8)));
typedef float  v16f  __attribute__((ext_vector_type(16)));

#define GRU_B 16
#define GRU_T 4096
#define GRU_D 256
#define GRU_N 256

#define WX_ELEMS (48 * 8 * 512)   // 48 col-tiles, 8 k-chunks, 512 bf16/tile
#define WH_ELEMS (48 * 8 * 512)
#define WPACK_BYTES ((size_t)(WX_ELEMS + WH_ELEMS) * 2)

__device__ __forceinline__ bf16_t f2bf(float f) { return (bf16_t)f; }

__device__ __forceinline__ v8f wmma_bf16(v16bf a, v16bf b, v8f c) {
  return __builtin_amdgcn_wmma_f32_16x16x32_bf16(
      false, a, false, b, (short)0, c, false, false);
}

__device__ __forceinline__ v16bf cat8(v8bf lo, v8bf hi) {
  return __builtin_shufflevector(lo, hi, 0, 1, 2, 3, 4, 5, 6, 7,
                                 8, 9, 10, 11, 12, 13, 14, 15);
}

__device__ __forceinline__ float fast_sigmoid(float x) {
  return 1.0f / (1.0f + __expf(-x));
}
__device__ __forceinline__ float fast_tanh(float x) {
  float e2 = __expf(2.0f * x);
  return 1.0f - 2.0f / (e2 + 1.0f);
}

// ---------------------------------------------------------------------------
// K0: pack weights.  Tile (ct, kc): 512 bf16, lane l owns 16 contiguous:
//   n  = ct*16 + (l&15);  kb = kc*32 + (l>=16 ? 8 : 0)
//   elems 0..7 = W[kb+0..7][n], elems 8..15 = W[kb+16..23][n]   (ISA 7.12.2)
// Tiles 0..383 = WX region (rows 0..255 of Wg / Wc), 384..767 = WH region
// (rows 256..511).  Within a region: ct<32 -> W_gates cols, ct>=32 -> W_cand.
// ---------------------------------------------------------------------------
__global__ void gru_pack_weights(const float* __restrict__ Wg,
                                 const float* __restrict__ Wc,
                                 bf16_t* __restrict__ packed) {
  int e = blockIdx.x * blockDim.x + threadIdx.x;
  if (e >= 768 * 512) return;
  int tile   = e >> 9;
  int within = e & 511;
  int lane   = within >> 4;
  int j      = within & 15;

  int region = (tile >= 384);
  int tl     = region ? tile - 384 : tile;
  int rowOff = region ? 256 : 0;
  int ct = tl >> 3;
  int kc = tl & 7;

  const float* src;
  int stride, n;
  if (ct < 32) { src = Wg; stride = 512; n = ct * 16 + (lane & 15); }
  else         { src = Wc; stride = 256; n = (ct - 32) * 16 + (lane & 15); }

  int kb = kc * 32 + ((lane >> 4) ? 8 : 0);
  int k  = kb + (j < 8 ? j : 16 + (j - 8));
  packed[e] = f2bf(src[(size_t)(rowOff + k) * stride + n]);
}

// ---------------------------------------------------------------------------
// K1: x-projection GEMM.  4096 blocks x 8 waves; wave w owns col-tiles
// c = 6w..6w+5 of the 48-tile fused column space.  Branchless B addressing.
// ---------------------------------------------------------------------------
__global__ void __launch_bounds__(256)
gru_xproj(const float* __restrict__ X,
          const float* __restrict__ bg, const float* __restrict__ bc,
          const bf16_t* __restrict__ WX,
          float* __restrict__ XG, float* __restrict__ out) {
  int mt  = blockIdx.x;
  int w   = threadIdx.x >> 5;
  int l   = threadIdx.x & 31;
  int am  = l & 15;
  int sel = l >> 4;

  const float* xr = X + ((size_t)mt * 16 + am) * GRU_D;

  v8f acc[6];
#pragma unroll
  for (int i = 0; i < 6; ++i) acc[i] = (v8f){};

#pragma unroll
  for (int kc = 0; kc < 8; ++kc) {
    int kb = kc * 32 + sel * 8;
    float4 q0 = *(const float4*)(xr + kb);
    float4 q1 = *(const float4*)(xr + kb + 4);
    float4 q2 = *(const float4*)(xr + kb + 16);
    float4 q3 = *(const float4*)(xr + kb + 20);
    v16f fx = {q0.x, q0.y, q0.z, q0.w, q1.x, q1.y, q1.z, q1.w,
               q2.x, q2.y, q2.z, q2.w, q3.x, q3.y, q3.z, q3.w};
    v16bf a = __builtin_convertvector(fx, v16bf);

#pragma unroll
    for (int i = 0; i < 6; ++i) {
      int c = w * 6 + i;
      v16bf b = *(const v16bf*)(WX + ((size_t)c * 8 + kc) * 512 + l * 16);
      acc[i] = wmma_bf16(a, b, acc[i]);
    }
  }

  // D layout: element (M = j + 8*sel, N = l&15) lives in acc[i][j]
#pragma unroll
  for (int i = 0; i < 6; ++i) {
    int c = w * 6 + i;
    if (c < 32) {                       // gates -> XG scratch
      int n = c * 16 + (l & 15);
      float bias = bg[n];
#pragma unroll
      for (int j = 0; j < 8; ++j) {
        size_t row = (size_t)mt * 16 + j + 8 * sel;   // row = b*T + t
        XG[row * 512 + n] = acc[i][j] + bias;
      }
    } else {                            // cand pre-activation -> d_out stash
      int n = (c - 32) * 16 + (l & 15);
      float bias = bc[n];
#pragma unroll
      for (int j = 0; j < 8; ++j) {
        size_t row = (size_t)mt * 16 + j + 8 * sel;
        out[row * GRU_N + n] = acc[i][j] + bias;
      }
    }
  }
}

// ---------------------------------------------------------------------------
// K2: sequential recurrence.  512 threads = 16 waves.
// ---------------------------------------------------------------------------
__global__ void __launch_bounds__(512)
gru_recurrence(const float* __restrict__ h0, const int* __restrict__ seq,
               const bf16_t* __restrict__ WH, const float* __restrict__ XG,
               float* __restrict__ out) {
  __shared__ float  hs [GRU_B][GRU_N];   // h, fp32 (update math)
  __shared__ bf16_t hsb[GRU_B][GRU_N];   // h, bf16 mirror (WMMA A)
  __shared__ bf16_t rhb[GRU_B][GRU_N];   // r*h, bf16 (WMMA A)
  __shared__ float  us [GRU_B][GRU_N];   // update gate
  __shared__ float  xgs[GRU_B][2 * GRU_N];  // staged XG slab for step t
  __shared__ float  xcs[GRU_B][GRU_N];      // staged Xc slab for step t
  __shared__ int    sl [GRU_B];

  int tid = threadIdx.x;
  for (int i = tid; i < GRU_B * GRU_N; i += 512) {
    float v = h0[i];
    (&hs[0][0])[i]  = v;
    (&hsb[0][0])[i] = f2bf(v);
  }
  if (tid < GRU_B) sl[tid] = seq[tid];
  __syncthreads();

  int w   = tid >> 5;      // wave 0..15
  int l   = tid & 31;
  int am  = l & 15;
  int sel = l >> 4;
  int ct0 = w * 2, ct1 = w * 2 + 1;   // gate col tiles (0..31)
  int nc  = w * 16 + (l & 15);        // this thread's candidate column

  for (int t = 0; t < GRU_T; ++t) {
    // ---- stage XG (16x512) and Xc (16x256) slabs, coalesced float4 ----
#pragma unroll
    for (int k = 0; k < 4; ++k) {
      int idx = tid + k * 512;               // 0..2047 float4s
      int m = idx >> 7, c4 = (idx & 127) * 4;
      const float* p = XG + ((size_t)m * GRU_T + t) * 512 + c4;
      *(float4*)&xgs[m][c4] = *(const float4*)p;
      if (t + 1 < GRU_T) __builtin_prefetch(p + 512, 0, 1);  // next step
    }
#pragma unroll
    for (int k = 0; k < 2; ++k) {
      int idx = tid + k * 512;               // 0..1023 float4s
      int m = idx >> 6, c4 = (idx & 63) * 4;
      const float* p = out + ((size_t)m * GRU_T + t) * 256 + c4;
      *(float4*)&xcs[m][c4] = *(const float4*)p;
      if (t + 1 < GRU_T) __builtin_prefetch(p + 256, 0, 1);
    }
    __syncthreads();                         // barrier 1

    // ---- gates GEMM: [16,256]@[256,512]; wave w owns tiles 2w, 2w+1 ----
    v8f acc0 = (v8f){}, acc1 = (v8f){};
#pragma unroll
    for (int kc = 0; kc < 8; ++kc) {
      int kb = kc * 32 + sel * 8;
      v16bf a = cat8(*(const v8bf*)&hsb[am][kb],
                     *(const v8bf*)&hsb[am][kb + 16]);
      v16bf b0 = *(const v16bf*)(WH + ((size_t)ct0 * 8 + kc) * 512 + l * 16);
      v16bf b1 = *(const v16bf*)(WH + ((size_t)ct1 * 8 + kc) * 512 + l * 16);
      acc0 = wmma_bf16(a, b0, acc0);
      acc1 = wmma_bf16(a, b1, acc1);
    }

    // pull xc into registers now so all xcs reads finish before barrier 2
    float xcr[8];
#pragma unroll
    for (int j = 0; j < 8; ++j) xcr[j] = xcs[j + 8 * sel][nc];

    // ---- sigmoid + r*h / u scatter to LDS ----
#pragma unroll
    for (int half = 0; half < 2; ++half) {
      v8f acc = half ? acc1 : acc0;
      int n = (half ? ct1 : ct0) * 16 + (l & 15);
#pragma unroll
      for (int j = 0; j < 8; ++j) {
        int mm = j + 8 * sel;
        float g = fast_sigmoid(acc[j] + xgs[mm][n]);
        if (n < 256) rhb[mm][n]       = f2bf(g * hs[mm][n]);
        else         us [mm][n - 256] = g;
      }
    }
    __syncthreads();                         // barrier 2

    // ---- candidate GEMM: [16,256]@[256,256]; wave w owns tile 32+w ----
    v8f accc = (v8f){};
#pragma unroll
    for (int kc = 0; kc < 8; ++kc) {
      int kb = kc * 32 + sel * 8;
      v16bf a = cat8(*(const v8bf*)&rhb[am][kb],
                     *(const v8bf*)&rhb[am][kb + 16]);
      v16bf b =
          *(const v16bf*)(WH + ((size_t)(32 + w) * 8 + kc) * 512 + l * 16);
      accc = wmma_bf16(a, b, accc);
    }

    // ---- tanh, state update, masked output ----
#pragma unroll
    for (int j = 0; j < 8; ++j) {
      int mm = j + 8 * sel;
      float c  = fast_tanh(accc[j] + xcr[j]);
      float u  = us[mm][nc];
      float hv = hs[mm][nc];
      float hn = u * hv + (1.0f - u) * c;
      bool active = (t < sl[mm]);
      float ho = active ? hn : hv;
      out[((size_t)mm * GRU_T + t) * GRU_N + nc] = active ? hn : 0.0f;
      hs[mm][nc]  = ho;
      hsb[mm][nc] = f2bf(ho);
    }
    // next iteration's staging writes are fenced from this step's reads by
    // barrier 2; h writes fenced from next gates reads by barrier 1.
  }

  __syncthreads();
  float* hf = out + (size_t)GRU_B * GRU_T * GRU_N;
  for (int i = tid; i < GRU_B * GRU_N; i += 512) hf[i] = (&hs[0][0])[i];
}

// ---------------------------------------------------------------------------
extern "C" void kernel_launch(void* const* d_in, const int* in_sizes, int n_in,
                              void* d_out, int out_size, void* d_ws, size_t ws_size,
                              hipStream_t stream) {
  const float* x   = (const float*)d_in[0];
  const float* h0  = (const float*)d_in[1];
  const int*   seq = (const int*)d_in[2];
  const float* Wg  = (const float*)d_in[3];
  const float* bg  = (const float*)d_in[4];
  const float* Wc  = (const float*)d_in[5];
  const float* bc  = (const float*)d_in[6];
  float* out = (float*)d_out;

  bf16_t* Wp = (bf16_t*)d_ws;
  const bf16_t* WX = Wp;
  const bf16_t* WH = Wp + WX_ELEMS;
  float* XG = (float*)((char*)d_ws + WPACK_BYTES);   // 65536 x 512 fp32

  (void)in_sizes; (void)n_in; (void)out_size; (void)ws_size;

  gru_pack_weights<<<(768 * 512 + 255) / 256, 256, 0, stream>>>(Wg, Wc, Wp);
  gru_xproj<<<4096, 256, 0, stream>>>(x, bg, bc, WX, XG, out);
  gru_recurrence<<<1, 512, 0, stream>>>(h0, seq, WH, XG, out);
}
